// GDNSequenceModel_21285857919523
// MI455X (gfx1250) — compile-verified, matrix operand order checked by reading, not run on hardware
//
#include <hip/hip_runtime.h>
#include <hip/hip_bf16.h>
#include <math.h>

#define B_    4
#define T_    2048
#define D_    1024
#define H_    4
#define DK_   256
#define DV_   512
#define KDIM_ 1024
#define VDIM_ 2048
#define FFN_  4096
#define NTOK_ (B_*T_)

typedef __attribute__((ext_vector_type(16))) _Float16 v16h;
typedef __attribute__((ext_vector_type(8)))  float    v8f;

union FragA { unsigned u[8]; v16h v; };

// Async global->LDS 16B per lane (CDNA5 GLOBAL_LOAD_ASYNC_TO_LDS_B128, ASYNCcnt).
#define ASYNC_TILE_LD(lds32, gptr)                                   \
    asm volatile("global_load_async_to_lds_b128 %0, %1, off"         \
                 :: "v"(lds32), "v"(gptr) : "memory")
#define WAIT_ASYNCCNT0() asm volatile("s_wait_asynccnt 0x0" ::: "memory")

__device__ __forceinline__ float warp_sum(float v) {
    #pragma unroll
    for (int m = 16; m > 0; m >>= 1) v += __shfl_xor(v, m, 32);
    return v;
}

// ---------------------------------------------------------------------------
// Weight convert: W [K,N] f32 -> Wt [N,K] f16 (transposed, K-major rows)
// ---------------------------------------------------------------------------
__global__ __launch_bounds__(256)
void wconv_t(const float* __restrict__ W, _Float16* __restrict__ Wt, int K, int N) {
    size_t id = (size_t)blockIdx.x * 256 + threadIdx.x;
    if (id >= (size_t)K * N) return;
    int k = (int)(id % K);
    int n = (int)(id / K);
    Wt[id] = (_Float16)W[(size_t)k * N + n];
}

// ---------------------------------------------------------------------------
// LayerNorm (wave per row) -> f16 output
// ---------------------------------------------------------------------------
__global__ __launch_bounds__(256)
void ln_f16(const float* __restrict__ x, const float* __restrict__ w,
            const float* __restrict__ b, _Float16* __restrict__ out, int dim) {
    int lane = threadIdx.x & 31, wid = threadIdx.x >> 5;
    size_t row = (size_t)blockIdx.x * 8 + wid;
    const float* xr = x + row * dim;
    float s = 0.f, s2 = 0.f;
    for (int j = lane; j < dim; j += 32) { float v = xr[j]; s += v; s2 += v * v; }
    s = warp_sum(s); s2 = warp_sum(s2);
    float mu  = s / dim;
    float var = s2 / dim - mu * mu;
    float inv = rsqrtf(var + 1e-5f);
    _Float16* orow = out + row * dim;
    for (int j = lane; j < dim; j += 32)
        orow[j] = (_Float16)((xr[j] - mu) * inv * w[j] + b[j]);
}

// ---------------------------------------------------------------------------
// Tiled WMMA GEMM: C[M,N] = A[M,K] @ Bt[N,K]^T, f16 inputs, f32 accum.
// Double-buffered LDS tiles filled by GLOBAL_LOAD_ASYNC_TO_LDS_B128.
// MODE 0: f32 store; 1: +resid; 2: +bias, exact GELU, f16 store; 3: +bias+resid
// Block 256 threads = 8 wave32; BM=BN=128, BK=32; wave computes 32x64 (2x4 tiles)
// ---------------------------------------------------------------------------
template<int MODE>
__global__ __launch_bounds__(256)
void gemm_wmma(const _Float16* __restrict__ A, const _Float16* __restrict__ Bt,
               float* __restrict__ Cf, _Float16* __restrict__ Ch,
               const float* __restrict__ bias, const float* __restrict__ resid,
               int N, int K) {
    __shared__ _Float16 As[2][128][32];
    __shared__ _Float16 Bs[2][128][32];
    const int tid  = threadIdx.x;
    const int lane = tid & 31;
    const int wid  = tid >> 5;
    const int wm   = wid >> 1;      // 0..3
    const int wn   = wid & 1;       // 0..1
    const int l16  = lane & 15;
    const int half = lane >> 4;     // 0/1: K sub-range select per ISA A/B layout
    const size_t bm = (size_t)blockIdx.x * 128;
    const size_t bn = (size_t)blockIdx.y * 128;

    v8f zero = {0.f,0.f,0.f,0.f,0.f,0.f,0.f,0.f};
    v8f acc[2][4];
    #pragma unroll
    for (int i = 0; i < 2; ++i)
        #pragma unroll
        for (int j = 0; j < 4; ++j) acc[i][j] = zero;

    const int ldRow = tid >> 2;          // 0..63
    const int seg   = (tid & 3) * 8;     // 0,8,16,24 halfs (16B aligned)

    // Per-thread global source pointers (2 rows of A tile, 2 rows of B tile).
    const _Float16* gA0 = A  + (bm + ldRow)      * K + seg;
    const _Float16* gA1 = A  + (bm + ldRow + 64) * K + seg;
    const _Float16* gB0 = Bt + (bn + ldRow)      * K + seg;
    const _Float16* gB1 = Bt + (bn + ldRow + 64) * K + seg;

    // LDS byte addresses (flat-pointer low 32 bits == LDS offset on CDNA5).
    unsigned lA0[2], lA1[2], lB0[2], lB1[2];
    #pragma unroll
    for (int b = 0; b < 2; ++b) {
        lA0[b] = (unsigned)(size_t)&As[b][ldRow][seg];
        lA1[b] = (unsigned)(size_t)&As[b][ldRow + 64][seg];
        lB0[b] = (unsigned)(size_t)&Bs[b][ldRow][seg];
        lB1[b] = (unsigned)(size_t)&Bs[b][ldRow + 64][seg];
    }

    // Prologue: async-fill buffer 0 with the first K tile.
    ASYNC_TILE_LD(lA0[0], gA0);
    ASYNC_TILE_LD(lA1[0], gA1);
    ASYNC_TILE_LD(lB0[0], gB0);
    ASYNC_TILE_LD(lB1[0], gB1);

    int buf = 0;
    for (int kb = 0; kb < K; kb += 32, buf ^= 1) {
        WAIT_ASYNCCNT0();       // own async fills done
        __syncthreads();        // waits dscnt then barrier: all waves' fills + prior reads done
        if (kb + 32 < K) {      // async-fill the other buffer while computing this one
            ASYNC_TILE_LD(lA0[buf ^ 1], gA0 + kb + 32);
            ASYNC_TILE_LD(lA1[buf ^ 1], gA1 + kb + 32);
            ASYNC_TILE_LD(lB0[buf ^ 1], gB0 + kb + 32);
            ASYNC_TILE_LD(lB1[buf ^ 1], gB1 + kb + 32);
        }
        if (kb + 64 < K) {      // warm GL2 for the tile after next
            __builtin_prefetch(gA0 + kb + 64, 0, 1);
            __builtin_prefetch(gB0 + kb + 64, 0, 1);
        }

        // Fragments per CDNA5 16-bit A 16x32 layout:
        // lane<16 holds K {0..7,16..23}, lane>=16 holds K {8..15,24..31}.
        v16h af[2], bf[4];
        #pragma unroll
        for (int mt = 0; mt < 2; ++mt) {
            int row = wm * 32 + mt * 16 + l16;
            FragA f;
            #pragma unroll
            for (int r = 0; r < 8; ++r) {
                int kk = ((r < 4) ? 2 * r : 8 + 2 * r) + half * 8;
                f.u[r] = *(const unsigned*)&As[buf][row][kk];
            }
            af[mt] = f.v;
        }
        #pragma unroll
        for (int nt = 0; nt < 4; ++nt) {
            int colr = wn * 64 + nt * 16 + l16;
            FragA f;
            #pragma unroll
            for (int r = 0; r < 8; ++r) {
                int kk = ((r < 4) ? 2 * r : 8 + 2 * r) + half * 8;
                f.u[r] = *(const unsigned*)&Bs[buf][colr][kk];
            }
            bf[nt] = f.v;
        }

        #pragma unroll
        for (int mt = 0; mt < 2; ++mt)
            #pragma unroll
            for (int nt = 0; nt < 4; ++nt)
                acc[mt][nt] = __builtin_amdgcn_wmma_f32_16x16x32_f16(
                    false, af[mt], false, bf[nt], (short)0, acc[mt][nt], false, false);
    }

    // Epilogue. C/D layout: VGPR r -> M = r + half*8 within tile, N = l16.
    #pragma unroll
    for (int mt = 0; mt < 2; ++mt) {
        #pragma unroll
        for (int nt = 0; nt < 4; ++nt) {
            #pragma unroll
            for (int r = 0; r < 8; ++r) {
                size_t row = bm + wm * 32 + mt * 16 + half * 8 + r;
                size_t col = bn + wn * 64 + nt * 16 + l16;
                size_t off = row * (size_t)N + col;
                float v = acc[mt][nt][r];
                if (MODE == 0) {
                    Cf[off] = v;
                } else if (MODE == 1) {
                    Cf[off] = v + resid[off];
                } else if (MODE == 2) {
                    v += bias[col];
                    v = 0.5f * v * (1.f + erff(v * 0.70710678118f));  // exact GELU
                    Ch[off] = (_Float16)v;
                } else {
                    Cf[off] = v + bias[col] + resid[off];
                }
            }
        }
    }
}

// ---------------------------------------------------------------------------
// Fused Wa/Wb projection -> g = -exp(A_log)*softplus(a+dt_bias), beta = sigmoid(b)
// One block per token; wave w handles head (w&3) of a (w<4) or b (w>=4).
// ---------------------------------------------------------------------------
__global__ __launch_bounds__(256)
void ab_proj(const _Float16* __restrict__ normed, const float* __restrict__ Wa,
             const float* __restrict__ Wb, const float* __restrict__ A_log,
             const float* __restrict__ dt_bias, float* __restrict__ g,
             float* __restrict__ beta) {
    int lane = threadIdx.x & 31, wid = threadIdx.x >> 5;
    int h = wid & 3;
    const float* W = (wid < 4) ? Wa : Wb;
    size_t row = blockIdx.x;
    const _Float16* nr = normed + row * D_;
    float acc = 0.f;
    for (int d = lane; d < D_; d += 32) acc += (float)nr[d] * W[(size_t)d * H_ + h];
    acc = warp_sum(acc);
    if (lane == 0) {
        if (wid < 4) {
            float z  = acc + dt_bias[h];
            float sp = (z > 20.f) ? z : log1pf(__expf(z));
            g[row * H_ + h] = -__expf(A_log[h]) * sp;
        } else {
            beta[row * H_ + h] = 1.f / (1.f + __expf(-acc));
        }
    }
}

// ---------------------------------------------------------------------------
// Causal depthwise conv (K=4) + SiLU
// ---------------------------------------------------------------------------
__global__ __launch_bounds__(256)
void conv_silu(const float* __restrict__ x, const float* __restrict__ w,
               float* __restrict__ y, int C) {
    size_t gid = (size_t)blockIdx.x * 256 + threadIdx.x;
    int c = (int)(gid % C);
    int t = (int)((gid / C) % T_);
    int b = (int)(gid / ((size_t)C * T_));
    float a = 0.f;
    #pragma unroll
    for (int i = 0; i < 4; ++i) {
        int tt = t - 3 + i;
        if (tt >= 0) a += x[(size_t)(b * T_ + tt) * C + c] * w[c * 4 + i];
    }
    y[gid] = a / (1.f + __expf(-a));   // SiLU
}

// ---------------------------------------------------------------------------
// L2 norm over rows of length DK_ (in place), wave per row
// ---------------------------------------------------------------------------
__global__ __launch_bounds__(256)
void l2norm_inplace(float* __restrict__ x, float scale) {
    int lane = threadIdx.x & 31, wid = threadIdx.x >> 5;
    size_t row = (size_t)blockIdx.x * 8 + wid;
    float* xr = x + row * DK_;
    float vals[DK_ / 32];
    float s2 = 0.f; int c = 0;
    for (int j = lane; j < DK_; j += 32) { float v = xr[j]; vals[c++] = v; s2 += v * v; }
    s2 = warp_sum(s2);
    float sc = rsqrtf(s2 + 1e-6f) * scale;
    c = 0;
    for (int j = lane; j < DK_; j += 32) xr[j] = vals[c++] * sc;
}

// ---------------------------------------------------------------------------
// Gated delta rule. One block per (b, h, DV-chunk of 128). 512 threads.
// State S[256][128] lives in registers: thread (rg, col) owns S[rg*64..+64][col].
// ---------------------------------------------------------------------------
__global__ __launch_bounds__(512)
void delta_rule(const float* __restrict__ q, const float* __restrict__ k,
                const float* __restrict__ v, const float* __restrict__ g,
                const float* __restrict__ beta, float* __restrict__ o) {
    __shared__ float k_s[DK_], q_s[DK_], v_s[128], red[512], delta_s[128];
    const int t   = threadIdx.x;
    const int col = t & 127;
    const int rg  = t >> 7;          // 0..3
    const int i0  = rg * 64;
    const int vc  = blockIdx.x & 3;
    const int h   = (blockIdx.x >> 2) & 3;
    const int b   = blockIdx.x >> 4;

    float s[64];
    #pragma unroll
    for (int r = 0; r < 64; ++r) s[r] = 0.f;

    for (int ts = 0; ts < T_; ++ts) {
        size_t rowBH  = (size_t)(b * T_ + ts) * H_ + h;
        size_t baseQK = rowBH * DK_;
        size_t baseV  = rowBH * DV_ + (size_t)vc * 128;
        if (t < DK_) k_s[t] = k[baseQK + t];
        else         q_s[t - DK_] = q[baseQK + (t - DK_)];
        if (t < 128) v_s[t] = v[baseV + t];
        float eg = __expf(g[rowBH]);
        float bb = beta[rowBH];
        __syncthreads();

        // mem_j = (sum_i k_i * S_old[i][j]) * eg  (decay folded outside the sum)
        float acc = 0.f;
        #pragma unroll
        for (int r = 0; r < 64; ++r) acc += k_s[i0 + r] * s[r];
        red[t] = acc;
        __syncthreads();
        if (t < 128) {
            float m = (red[t] + red[128 + t] + red[256 + t] + red[384 + t]) * eg;
            delta_s[t] = (v_s[t] - m) * bb;
        }
        __syncthreads();

        // S = S*eg + k (x) delta ; o_j = sum_i q_i * S_new[i][j]
        float dl = delta_s[col];
        float oacc = 0.f;
        #pragma unroll
        for (int r = 0; r < 64; ++r) {
            float sv = s[r] * eg + k_s[i0 + r] * dl;
            s[r] = sv;
            oacc += q_s[i0 + r] * sv;
        }
        red[t] = oacc;
        __syncthreads();
        if (t < 128)
            o[baseV + t] = red[t] + red[128 + t] + red[256 + t] + red[384 + t];
        __syncthreads();
    }
}

// ---------------------------------------------------------------------------
// Gated RMSNorm: o * rsqrt(mean(o^2)+eps) * w * silu(gate) -> f16
// Wave per (b,t,h) row of DV_.
// ---------------------------------------------------------------------------
__global__ __launch_bounds__(256)
void gated_rms(const float* __restrict__ o, const float* __restrict__ gate,
               const float* __restrict__ w, _Float16* __restrict__ out) {
    int lane = threadIdx.x & 31, wid = threadIdx.x >> 5;
    size_t row = (size_t)blockIdx.x * 8 + wid;
    const float* orow = o + row * DV_;
    const float* grow = gate + row * DV_;
    float vals[DV_ / 32];
    float s2 = 0.f; int c = 0;
    for (int j = lane; j < DV_; j += 32) { float v = orow[j]; vals[c++] = v; s2 += v * v; }
    s2 = warp_sum(s2);
    float rms = rsqrtf(s2 / DV_ + 1e-5f);
    c = 0;
    for (int j = lane; j < DV_; j += 32) {
        float gt = grow[j];
        float sg = gt / (1.f + __expf(-gt));
        out[row * DV_ + j] = (_Float16)(vals[c++] * rms * w[j] * sg);
    }
}

// ---------------------------------------------------------------------------
extern "C" void kernel_launch(void* const* d_in, const int* in_sizes, int n_in,
                              void* d_out, int out_size, void* d_ws, size_t ws_size,
                              hipStream_t stream) {
    (void)in_sizes; (void)n_in; (void)out_size; (void)ws_size;
    const float* x     = (const float*)d_in[0];
    const float* Wq    = (const float*)d_in[1];
    const float* Wk    = (const float*)d_in[2];
    const float* Wv    = (const float*)d_in[3];
    const float* Wa    = (const float*)d_in[4];
    const float* Wb    = (const float*)d_in[5];
    const float* Wg    = (const float*)d_in[6];
    const float* cqw   = (const float*)d_in[7];
    const float* ckw   = (const float*)d_in[8];
    const float* cvw   = (const float*)d_in[9];
    const float* A_log = (const float*)d_in[10];
    const float* dtb   = (const float*)d_in[11];
    const float* onw   = (const float*)d_in[12];
    const float* Wo    = (const float*)d_in[13];
    const float* ln1w  = (const float*)d_in[14];
    const float* ln1b  = (const float*)d_in[15];
    const float* ln2w  = (const float*)d_in[16];
    const float* ln2b  = (const float*)d_in[17];
    const float* fw1   = (const float*)d_in[18];
    const float* fb1   = (const float*)d_in[19];
    const float* fw2   = (const float*)d_in[20];
    const float* fb2   = (const float*)d_in[21];
    float* out = (float*)d_out;

    char* p = (char*)d_ws;
    auto alloc = [&](size_t bytes) -> void* {
        void* r = (void*)p;
        p += (bytes + 255) & ~(size_t)255;
        return r;
    };

    _Float16* normed_h = (_Float16*)alloc((size_t)NTOK_ * D_ * 2);
    _Float16* WqT = (_Float16*)alloc((size_t)KDIM_ * D_ * 2);
    _Float16* WkT = (_Float16*)alloc((size_t)KDIM_ * D_ * 2);
    _Float16* WvT = (_Float16*)alloc((size_t)VDIM_ * D_ * 2);
    _Float16* WgT = (_Float16*)alloc((size_t)VDIM_ * D_ * 2);
    _Float16* WoT = (_Float16*)alloc((size_t)D_ * VDIM_ * 2);
    _Float16* W1T = (_Float16*)alloc((size_t)FFN_ * D_ * 2);
    _Float16* W2T = (_Float16*)alloc((size_t)D_ * FFN_ * 2);
    float* q_lin    = (float*)alloc((size_t)NTOK_ * KDIM_ * 4);
    float* k_lin    = (float*)alloc((size_t)NTOK_ * KDIM_ * 4);
    float* v_lin    = (float*)alloc((size_t)NTOK_ * VDIM_ * 4);
    float* gate_lin = (float*)alloc((size_t)NTOK_ * VDIM_ * 4);
    float* q_conv   = (float*)alloc((size_t)NTOK_ * KDIM_ * 4);
    float* k_conv   = (float*)alloc((size_t)NTOK_ * KDIM_ * 4);
    float* v_conv   = (float*)alloc((size_t)NTOK_ * VDIM_ * 4);
    float* gvec     = (float*)alloc((size_t)NTOK_ * H_ * 4);
    float* bvec     = (float*)alloc((size_t)NTOK_ * H_ * 4);
    // Safe aliases (producer finishes before consumer reuses):
    float*    o_buf  = v_lin;                 // o [B,T,H,DV]  (v_lin dead after conv_v)
    float*    x2     = q_lin;                 // x + o@Wo      (q_lin dead after conv_q)
    _Float16* h_h    = (_Float16*)k_lin;      // LN2 out       (k_lin dead after conv_k)
    _Float16* ffn1_h = (_Float16*)gate_lin;   // GELU out f16  (gate dead after gated_rms)
    _Float16* og_h   = (_Float16*)k_conv;     // gated-RMS out (k_conv dead after delta)

    dim3 blk(256);
    int ne;
    ne = D_ * KDIM_; wconv_t<<<(ne + 255) / 256, blk, 0, stream>>>(Wq, WqT, D_, KDIM_);
    wconv_t<<<(ne + 255) / 256, blk, 0, stream>>>(Wk, WkT, D_, KDIM_);
    ne = D_ * VDIM_; wconv_t<<<(ne + 255) / 256, blk, 0, stream>>>(Wv, WvT, D_, VDIM_);
    wconv_t<<<(ne + 255) / 256, blk, 0, stream>>>(Wg, WgT, D_, VDIM_);
    ne = VDIM_ * D_; wconv_t<<<(ne + 255) / 256, blk, 0, stream>>>(Wo, WoT, VDIM_, D_);
    ne = D_ * FFN_;  wconv_t<<<(ne + 255) / 256, blk, 0, stream>>>(fw1, W1T, D_, FFN_);
    ne = FFN_ * D_;  wconv_t<<<(ne + 255) / 256, blk, 0, stream>>>(fw2, W2T, FFN_, D_);

    ln_f16<<<NTOK_ / 8, blk, 0, stream>>>(x, ln1w, ln1b, normed_h, D_);

    gemm_wmma<0><<<dim3(NTOK_ / 128, KDIM_ / 128), blk, 0, stream>>>(
        normed_h, WqT, q_lin, nullptr, nullptr, nullptr, KDIM_, D_);
    gemm_wmma<0><<<dim3(NTOK_ / 128, KDIM_ / 128), blk, 0, stream>>>(
        normed_h, WkT, k_lin, nullptr, nullptr, nullptr, KDIM_, D_);
    gemm_wmma<0><<<dim3(NTOK_ / 128, VDIM_ / 128), blk, 0, stream>>>(
        normed_h, WvT, v_lin, nullptr, nullptr, nullptr, VDIM_, D_);
    gemm_wmma<0><<<dim3(NTOK_ / 128, VDIM_ / 128), blk, 0, stream>>>(
        normed_h, WgT, gate_lin, nullptr, nullptr, nullptr, VDIM_, D_);

    ab_proj<<<NTOK_, blk, 0, stream>>>(normed_h, Wa, Wb, A_log, dtb, gvec, bvec);

    conv_silu<<<(NTOK_ * KDIM_) / 256, blk, 0, stream>>>(q_lin, cqw, q_conv, KDIM_);
    conv_silu<<<(NTOK_ * KDIM_) / 256, blk, 0, stream>>>(k_lin, ckw, k_conv, KDIM_);
    conv_silu<<<(NTOK_ * VDIM_) / 256, blk, 0, stream>>>(v_lin, cvw, v_conv, VDIM_);

    l2norm_inplace<<<(NTOK_ * H_) / 8, blk, 0, stream>>>(q_conv, 0.0625f);  // * DK^-0.5
    l2norm_inplace<<<(NTOK_ * H_) / 8, blk, 0, stream>>>(k_conv, 1.0f);

    delta_rule<<<B_ * H_ * 4, 512, 0, stream>>>(q_conv, k_conv, v_conv, gvec, bvec, o_buf);

    gated_rms<<<(NTOK_ * H_) / 8, blk, 0, stream>>>(o_buf, gate_lin, onw, og_h);

    gemm_wmma<1><<<dim3(NTOK_ / 128, D_ / 128), blk, 0, stream>>>(
        og_h, WoT, x2, nullptr, nullptr, x, D_, VDIM_);

    ln_f16<<<NTOK_ / 8, blk, 0, stream>>>(x2, ln2w, ln2b, h_h, D_);

    gemm_wmma<2><<<dim3(NTOK_ / 128, FFN_ / 128), blk, 0, stream>>>(
        h_h, W1T, nullptr, ffn1_h, fb1, nullptr, FFN_, D_);

    gemm_wmma<3><<<dim3(NTOK_ / 128, D_ / 128), blk, 0, stream>>>(
        ffn1_h, W2T, out, nullptr, fb2, x2, D_, FFN_);
}
